// SelfAttention_41429254537370
// MI455X (gfx1250) — matrix operand order repaired and checked by reading, not compiled, not run on hardware
//
#include <hip/hip_runtime.h>
#include <hip/hip_bf16.h>

typedef _Float16 half_t;
typedef __attribute__((ext_vector_type(16))) _Float16 v16h;
typedef __attribute__((ext_vector_type(8)))  _Float16 v8h;
typedef __attribute__((ext_vector_type(8)))  float    v8f;

#define B_  2
#define S_  2048
#define D_  1024
#define H_  16
#define HD_ 64
#define N3_ 3072

static __device__ __forceinline__ v16h cat8(v8h lo, v8h hi) {
  return __builtin_shufflevector(lo, hi, 0,1,2,3,4,5,6,7,8,9,10,11,12,13,14,15);
}
static __device__ __forceinline__ v8f wmma_f16(v16h a, v16h b, v8f c) {
  // D(16x16 f32) = A(16x32 f16) * B(32x16 f16) + C
  return __builtin_amdgcn_wmma_f32_16x16x32_f16(false, a, false, b, (short)0, c,
                                                false, false);
}

// Async global -> LDS copy (CDNA5 GLOBAL_LOAD_ASYNC_TO_LDS_B128).
// Per ISA 15.18.3 the instruction offset is added to both the LDS and the global address,
// so one address pair serves both halves via the instruction offset.
// LDS byte address = low 32 bits of the generic shared-space pointer.
static __device__ __forceinline__ void async_cp16(const half_t* g, half_t* l) {
  unsigned la = (unsigned)(unsigned long long)l;
  asm volatile("global_load_async_to_lds_b128 %0, %1, off"
               :: "v"(la), "v"(g) : "memory");
}
static __device__ __forceinline__ void async_cp32(const half_t* g, half_t* l) {
  unsigned la = (unsigned)(unsigned long long)l;
  asm volatile("global_load_async_to_lds_b128 %0, %1, off"
               :: "v"(la), "v"(g) : "memory");
  asm volatile("global_load_async_to_lds_b128 %0, %1, off offset:16"
               :: "v"(la), "v"(g) : "memory");
}
static __device__ __forceinline__ void wait_async0() {
  asm volatile("s_wait_asynccnt 0x0" ::: "memory");
}

// ---------------------------------------------------------------- convert
__global__ void cvt_f32_f16(const float* __restrict__ in,
                            half_t* __restrict__ out, int n) {
  int i = blockIdx.x * blockDim.x + threadIdx.x;
  int st = gridDim.x * blockDim.x;
  for (; i < n; i += st) out[i] = (half_t)in[i];
}

// ---------------------------------------------------------------- QKV GEMM
// qkv[m, n] = sum_k xh[m,k] * w_in[n,k] + b_in[n];  M=4096, N=3072, K=1024
// Double-buffered LDS, async global->LDS staging overlapped with WMMA.
__global__ __launch_bounds__(256) void qkv_gemm(
    const half_t* __restrict__ xh, const half_t* __restrict__ wh,
    const float* __restrict__ b_in, half_t* __restrict__ Qh,
    half_t* __restrict__ Kh, half_t* __restrict__ Vh) {
  __shared__ half_t sA[2][128 * 40];
  __shared__ half_t sB[2][128 * 40];
  const int tid  = threadIdx.x;
  const int lane = tid & 31;
  const int wid  = tid >> 5;
  const int hi   = (lane >> 4) & 1;
  const int ln   = lane & 15;
  const int mBase = blockIdx.y * 128;
  const int nBase = blockIdx.x * 128;
  const int waveM = wid >> 1;  // 0..3 -> 32 rows each
  const int waveN = wid & 1;   // 0..1 -> 64 cols each

  v8f acc[2][4];
#pragma unroll
  for (int a = 0; a < 2; a++)
#pragma unroll
    for (int b = 0; b < 4; b++) acc[a][b] = v8f{};

  const int ldRow = tid >> 1;
  const int ldCol = (tid & 1) * 16;
  const half_t* gA = xh + (mBase + ldRow) * 1024 + ldCol;
  const half_t* gB = wh + (nBase + ldRow) * 1024 + ldCol;
  const int lOff = ldRow * 40 + ldCol;

  // prologue: stage k-tile 0 into buffer 0
  async_cp32(gA, &sA[0][lOff]);
  async_cp32(gB, &sB[0][lOff]);

  for (int it = 0; it < 32; it++) {
    const int cur = it & 1;
    wait_async0();        // own async stores into sA/sB[cur] complete
    __syncthreads();      // everyone's done (and done reading the other buf)
    if (it + 1 < 32) {
      const int k0 = (it + 1) * 32;
      async_cp32(gA + k0, &sA[1 - cur][lOff]);
      async_cp32(gB + k0, &sB[1 - cur][lOff]);
    }

    v16h af[2], bf[4];
#pragma unroll
    for (int tm = 0; tm < 2; tm++) {
      int m = waveM * 32 + tm * 16 + ln;
      af[tm] = cat8(*(const v8h*)&sA[cur][m * 40 + hi * 8],
                    *(const v8h*)&sA[cur][m * 40 + 16 + hi * 8]);
    }
#pragma unroll
    for (int tn = 0; tn < 4; tn++) {
      int n = waveN * 64 + tn * 16 + ln;
      bf[tn] = cat8(*(const v8h*)&sB[cur][n * 40 + hi * 8],
                    *(const v8h*)&sB[cur][n * 40 + 16 + hi * 8]);
    }
#pragma unroll
    for (int tm = 0; tm < 2; tm++)
#pragma unroll
      for (int tn = 0; tn < 4; tn++)
        acc[tm][tn] = wmma_f16(af[tm], bf[tn], acc[tm][tn]);
  }

  // epilogue: add bias, scatter to Q/K/V [B,H,S,HD] as f16
#pragma unroll
  for (int tn = 0; tn < 4; tn++) {
    int gCol = nBase + waveN * 64 + tn * 16 + ln;  // 0..3071
    float bias = b_in[gCol];
    int which = gCol >> 10;       // 0:Q 1:K 2:V
    int c = gCol & 1023;
    int h = c >> 6, hd = c & 63;
    half_t* dst = (which == 0) ? Qh : ((which == 1) ? Kh : Vh);
#pragma unroll
    for (int tm = 0; tm < 2; tm++) {
#pragma unroll
      for (int j = 0; j < 8; j++) {
        int gRow = mBase + waveM * 32 + tm * 16 + hi * 8 + j;  // 0..4095
        int b = gRow >> 11, s = gRow & 2047;
        float v = acc[tm][tn][j] + bias;
        dst[(((b * H_ + h) * S_ + s) * HD_) + hd] = (half_t)v;
      }
    }
  }
}

// ---------------------------------------------------------------- attention
// One block = 128 query rows of one (b,h). 8 waves x 16 rows. Flash-style
// streaming softmax over 32-key tiles. K tile staged via async copy.
__global__ __launch_bounds__(256) void attn_kernel(
    const half_t* __restrict__ Qh, const half_t* __restrict__ Kh,
    const half_t* __restrict__ Vh, half_t* __restrict__ Oh) {
  __shared__ half_t sK[32 * 72];       // [key][dim], padded stride
  __shared__ half_t sVt[64 * 40];      // [dim][key], transposed, padded
  __shared__ half_t sP[8 * 16 * 40];   // per-wave P tile [16 rows][32 keys]
  const int tid = threadIdx.x, lane = tid & 31, wid = tid >> 5;
  const int hi = (lane >> 4) & 1, ln = lane & 15;
  const int h = blockIdx.y, b = blockIdx.z;
  const int bh = b * H_ + h;
  const half_t* Qp = Qh + bh * (S_ * HD_);
  const half_t* Kp = Kh + bh * (S_ * HD_);
  const half_t* Vp = Vh + bh * (S_ * HD_);
  const int q0 = blockIdx.x * 128 + wid * 16;

  // Q fragments (A layout), pre-scaled by 1/sqrt(64)=0.125
  v16h qf[2];
#pragma unroll
  for (int c = 0; c < 2; c++) {
    const half_t* g = Qp + (q0 + ln) * HD_ + c * 32 + hi * 8;
    v16h t = cat8(*(const v8h*)g, *(const v8h*)(g + 16));
#pragma unroll
    for (int i = 0; i < 16; i++) t[i] = t[i] * (_Float16)0.125f;
    qf[c] = t;
  }

  v8f outa[4];
#pragma unroll
  for (int t = 0; t < 4; t++) outa[t] = v8f{};
  float row_m[8], row_l[8];
#pragma unroll
  for (int j = 0; j < 8; j++) { row_m[j] = -1e30f; row_l[j] = 0.0f; }

  half_t* myP = &sP[wid * 16 * 40];
  const int kkey = tid >> 3;        // 0..31
  const int kdim = (tid & 7) * 8;   // 0..56

  for (int kt = 0; kt < S_ / 32; kt++) {
    __syncthreads();
    {
      // K tile: async copy straight into LDS (8 halfs per thread)
      async_cp16(Kp + (kt * 32 + kkey) * HD_ + kdim, &sK[kkey * 72 + kdim]);
      // V tile: transpose through registers
      v8h vv = *(const v8h*)(Vp + (kt * 32 + kkey) * HD_ + kdim);
#pragma unroll
      for (int i = 0; i < 8; i++) sVt[(kdim + i) * 40 + kkey] = vv[i];
    }
    wait_async0();
    __syncthreads();

    // scores: s0 = Q*K^T (keys 0..15 of tile), s1 = keys 16..31
    v8f s0 = v8f{}, s1 = v8f{};
#pragma unroll
    for (int c = 0; c < 2; c++) {
      v16h kb0 = cat8(*(const v8h*)&sK[ln * 72 + c * 32 + hi * 8],
                      *(const v8h*)&sK[ln * 72 + c * 32 + 16 + hi * 8]);
      v16h kb1 = cat8(*(const v8h*)&sK[(16 + ln) * 72 + c * 32 + hi * 8],
                      *(const v8h*)&sK[(16 + ln) * 72 + c * 32 + 16 + hi * 8]);
      s0 = wmma_f16(qf[c], kb0, s0);
      s1 = wmma_f16(qf[c], kb1, s1);
    }

    // online softmax update (row = hi*8+j; the 16 cols of a row live in the
    // 16 lanes of this half-wave -> shfl_xor masks 1,2,4,8)
#pragma unroll
    for (int j = 0; j < 8; j++) {
      float tmax = fmaxf(s0[j], s1[j]);
#pragma unroll
      for (int m = 1; m < 16; m <<= 1) tmax = fmaxf(tmax, __shfl_xor(tmax, m, 32));
      float nm = fmaxf(row_m[j], tmax);
      float alpha = __expf(row_m[j] - nm);
      float p0 = __expf(s0[j] - nm);
      float p1 = __expf(s1[j] - nm);
      float ts = p0 + p1;
#pragma unroll
      for (int m = 1; m < 16; m <<= 1) ts += __shfl_xor(ts, m, 32);
      row_l[j] = row_l[j] * alpha + ts;
      row_m[j] = nm;
#pragma unroll
      for (int t = 0; t < 4; t++) outa[t][j] *= alpha;
      int r = hi * 8 + j;
      myP[r * 40 + ln] = (half_t)p0;
      myP[r * 40 + 16 + ln] = (half_t)p1;
    }
    __syncthreads();

    // P (16x32) * V (32x64): A-fragment of P from LDS, B-fragments from sVt
    v16h pf = cat8(*(const v8h*)&myP[ln * 40 + hi * 8],
                   *(const v8h*)&myP[ln * 40 + 16 + hi * 8]);
#pragma unroll
    for (int t = 0; t < 4; t++) {
      v16h vb = cat8(*(const v8h*)&sVt[(t * 16 + ln) * 40 + hi * 8],
                     *(const v8h*)&sVt[(t * 16 + ln) * 40 + 16 + hi * 8]);
      outa[t] = wmma_f16(pf, vb, outa[t]);
    }
  }

  // finalize: divide by row sums, write f16 to Oh [B,S,D]
#pragma unroll
  for (int j = 0; j < 8; j++) {
    float inv = 1.0f / row_l[j];
    int s = q0 + hi * 8 + j;
#pragma unroll
    for (int t = 0; t < 4; t++) {
      float v = outa[t][j] * inv;
      Oh[(b * S_ + s) * D_ + h * HD_ + t * 16 + ln] = (half_t)v;
    }
  }
}

// ---------------------------------------------------------------- out proj
// out[m, n] = sum_k Oh[m,k] * w_out[n,k] + b_out[n];  M=4096, N=1024, K=1024
__global__ __launch_bounds__(256) void out_gemm(
    const half_t* __restrict__ Ah, const half_t* __restrict__ wh,
    const float* __restrict__ b_out, float* __restrict__ out) {
  __shared__ half_t sA[2][128 * 40];
  __shared__ half_t sB[2][128 * 40];
  const int tid  = threadIdx.x;
  const int lane = tid & 31;
  const int wid  = tid >> 5;
  const int hi   = (lane >> 4) & 1;
  const int ln   = lane & 15;
  const int mBase = blockIdx.y * 128;
  const int nBase = blockIdx.x * 128;
  const int waveM = wid >> 1;
  const int waveN = wid & 1;

  v8f acc[2][4];
#pragma unroll
  for (int a = 0; a < 2; a++)
#pragma unroll
    for (int b = 0; b < 4; b++) acc[a][b] = v8f{};

  const int ldRow = tid >> 1;
  const int ldCol = (tid & 1) * 16;
  const half_t* gA = Ah + (mBase + ldRow) * 1024 + ldCol;
  const half_t* gB = wh + (nBase + ldRow) * 1024 + ldCol;
  const int lOff = ldRow * 40 + ldCol;

  async_cp32(gA, &sA[0][lOff]);
  async_cp32(gB, &sB[0][lOff]);

  for (int it = 0; it < 32; it++) {
    const int cur = it & 1;
    wait_async0();
    __syncthreads();
    if (it + 1 < 32) {
      const int k0 = (it + 1) * 32;
      async_cp32(gA + k0, &sA[1 - cur][lOff]);
      async_cp32(gB + k0, &sB[1 - cur][lOff]);
    }

    v16h af[2], bf[4];
#pragma unroll
    for (int tm = 0; tm < 2; tm++) {
      int m = waveM * 32 + tm * 16 + ln;
      af[tm] = cat8(*(const v8h*)&sA[cur][m * 40 + hi * 8],
                    *(const v8h*)&sA[cur][m * 40 + 16 + hi * 8]);
    }
#pragma unroll
    for (int tn = 0; tn < 4; tn++) {
      int n = waveN * 64 + tn * 16 + ln;
      bf[tn] = cat8(*(const v8h*)&sB[cur][n * 40 + hi * 8],
                    *(const v8h*)&sB[cur][n * 40 + 16 + hi * 8]);
    }
#pragma unroll
    for (int tm = 0; tm < 2; tm++)
#pragma unroll
      for (int tn = 0; tn < 4; tn++)
        acc[tm][tn] = wmma_f16(af[tm], bf[tn], acc[tm][tn]);
  }

#pragma unroll
  for (int tn = 0; tn < 4; tn++) {
    int gCol = nBase + waveN * 64 + tn * 16 + ln;
    float bias = b_out[gCol];
#pragma unroll
    for (int tm = 0; tm < 2; tm++) {
#pragma unroll
      for (int j = 0; j < 8; j++) {
        int gRow = mBase + waveM * 32 + tm * 16 + hi * 8 + j;
        out[gRow * 1024 + gCol] = acc[tm][tn][j] + bias;
      }
    }
  }
}

// ---------------------------------------------------------------- launcher
extern "C" void kernel_launch(void* const* d_in, const int* in_sizes, int n_in,
                              void* d_out, int out_size, void* d_ws,
                              size_t ws_size, hipStream_t stream) {
  const float* x     = (const float*)d_in[0];
  const float* w_in  = (const float*)d_in[1];
  const float* b_in  = (const float*)d_in[2];
  const float* w_out = (const float*)d_in[3];
  const float* b_out = (const float*)d_in[4];
  float* out = (float*)d_out;

  half_t* ws = (half_t*)d_ws;
  half_t* xh    = ws;                        // 4 Mi halfs
  half_t* winh  = ws + (4u << 20);           // 3 Mi halfs
  half_t* wouth = ws + (7u << 20);           // 1 Mi halfs
  half_t* Qh    = ws + (8u << 20);           // 4 Mi halfs
  half_t* Kh    = ws + (12u << 20);          // 4 Mi halfs
  half_t* Vh    = ws + (16u << 20);          // 4 Mi halfs
  half_t* Oh    = ws + (20u << 20);          // 4 Mi halfs  (48 MB total)

  cvt_f32_f16<<<1024, 256, 0, stream>>>(x, xh, B_ * S_ * D_);
  cvt_f32_f16<<<1024, 256, 0, stream>>>(w_in, winh, N3_ * D_);
  cvt_f32_f16<<<512, 256, 0, stream>>>(w_out, wouth, D_ * D_);

  qkv_gemm<<<dim3(N3_ / 128, (B_ * S_) / 128), 256, 0, stream>>>(
      xh, winh, b_in, Qh, Kh, Vh);

  attn_kernel<<<dim3(S_ / 128, H_, B_), 256, 0, stream>>>(Qh, Kh, Vh, Oh);

  out_gemm<<<dim3(D_ / 128, (B_ * S_) / 128), 256, 0, stream>>>(
      Oh, wouth, b_out, out);
}
